// TimeVarKP_87299505259270
// MI455X (gfx1250) — matrix operand order repaired and checked by reading, not compiled, not run on hardware
//
#include <hip/hip_runtime.h>

// ---------------------------------------------------------------------------
// Problem constants (from reference)
// ---------------------------------------------------------------------------
#define Bz    512
#define FREQ  8
#define SEG   48
#define ENC   64
#define DIN   3072          // SEG*ENC
#define Edim  1024
#define Hh    8
#define HD    128           // Edim/Hh
#define Lrows 512           // attention sequence length (= Bz)
#define MROWS (Bz * FREQ)   // 4096 activation rows
#define STEP  4

typedef unsigned short u16t;
typedef unsigned int   u32t;

typedef __attribute__((ext_vector_type(16))) __bf16 v16bf;
typedef __attribute__((ext_vector_type(8)))  float  v8f;
typedef __attribute__((ext_vector_type(4)))  int    v4i_t;

struct U256 { uint4 lo; uint4 hi; };   // 32 bytes, 16B-aligned

// ---- CDNA5 async global->LDS copy (ASYNCcnt-tracked), with fallback --------
#if defined(__has_builtin) && __has_builtin(__builtin_amdgcn_global_load_async_to_lds_b128)
#define HAVE_ASYNC_COPY 1
#else
#define HAVE_ASYNC_COPY 0
#endif

typedef __attribute__((address_space(1))) v4i_t gv4i_t;   // global int4
typedef __attribute__((address_space(3))) v4i_t lv4i_t;   // LDS int4

static __device__ inline void async_copy16(const u16t* g, u16t* l) {
#if HAVE_ASYNC_COPY
    __builtin_amdgcn_global_load_async_to_lds_b128((gv4i_t*)(const void*)g,
                                                   (lv4i_t*)(void*)l, 0, 0);
#else
    *reinterpret_cast<uint4*>(l) = *reinterpret_cast<const uint4*>(g);
#endif
}

template <int N>
static __device__ inline void async_wait() {
#if HAVE_ASYNC_COPY
#if defined(__has_builtin) && __has_builtin(__builtin_amdgcn_s_wait_asynccnt)
    __builtin_amdgcn_s_wait_asynccnt(N);
#else
    asm volatile("s_wait_asynccnt %0" ::"i"(N));
#endif
#endif
}

static __device__ inline u16t f2bf(float f) {
    u32t u = __builtin_bit_cast(u32t, f);
    u32t r = u + 0x7FFFu + ((u >> 16) & 1u);   // round-to-nearest-even
    return (u16t)(r >> 16);
}

// Load a 16-element bf16 fragment as two 16-byte chunks (p0, p1 must be 16B aligned)
static __device__ inline v16bf load_frag(const u16t* p0, const u16t* p1) {
    U256 u;
    u.lo = *reinterpret_cast<const uint4*>(p0);
    u.hi = *reinterpret_cast<const uint4*>(p1);
    return __builtin_bit_cast(v16bf, u);
}

static __device__ inline v8f wmma_bf16(v16bf a, v16bf b, v8f c) {
    return __builtin_amdgcn_wmma_f32_16x16x32_bf16(
        /*neg_a=*/false, a, /*neg_b=*/false, b,
        /*c_mod=*/(short)0, c, /*reuse_a=*/false, /*reuse_b=*/false);
}

// ---------------------------------------------------------------------------
// fp32 -> bf16 converts
// ---------------------------------------------------------------------------
__global__ void convert_bf16_kernel(const float* __restrict__ in,
                                    u16t* __restrict__ out, long n) {
    long i = (long)blockIdx.x * blockDim.x + threadIdx.x;
    if (i < n) out[i] = f2bf(in[i]);
}

// in: R x C (row-major f32) -> out: C x R (row-major bf16), out[c*R+r] = in[r*C+c]
__global__ void transpose_bf16_kernel(const float* __restrict__ in,
                                      u16t* __restrict__ out, int R, int C) {
    long i = (long)blockIdx.x * blockDim.x + threadIdx.x;
    long total = (long)R * C;
    if (i < total) {
        long c = i / R, r = i - c * R;
        out[i] = f2bf(in[r * (long)C + c]);
    }
}

// Build V^T: Vt[(n*8+h)][d][m] = QKV[(m*8+n)][2048 + h*128 + d]
__global__ void build_vt_kernel(const u16t* __restrict__ QKV,
                                u16t* __restrict__ Vt) {
    long i = (long)blockIdx.x * blockDim.x + threadIdx.x;   // 64*128*512 total
    if (i >= (long)64 * HD * Lrows) return;
    int m  = (int)(i & 511);
    int d  = (int)((i >> 9) & 127);
    int nh = (int)(i >> 16);
    int n = nh >> 3, h = nh & 7;
    Vt[i] = QKV[((long)(m * FREQ + n)) * DIN + 2 * Edim + h * HD + d];
}

// ---------------------------------------------------------------------------
// Generic bf16 WMMA GEMM: C(MxN) = A(MxK) * Bt^T + bias
//   A  : M x K bf16 row-major, row stride lda (kept in global; reused 4x in regs)
//   Bt : N x K bf16 row-major; the 64x32 block tile is staged to LDS with
//        async global->LDS copies, double-buffered (copy k+1 overlaps WMMA k)
//   outF (f32, optional, non-temporal): row stride ldcF, replicated repCount
//        times at repStride offsets (x_pred replication)
//   outB (bf16, optional): row stride ldcB
// Block: 128 threads (4 waves), tile 64(M) x 64(N); wave w -> rows w*16..+15
// ---------------------------------------------------------------------------
#define BPITCH 40   // LDS row pitch (elements): 16 distinct banks for frag reads

__global__ __launch_bounds__(128) void gemm_bf16_wmma(
    const u16t* __restrict__ A, long lda,
    const u16t* __restrict__ Bt,
    const float* __restrict__ bias,
    float* __restrict__ outF, long ldcF, int repCount, long repStride,
    u16t* __restrict__ outB, long ldcB,
    int K)
{
    __shared__ __align__(16) u16t sB[2][64 * BPITCH];

    const int tid  = threadIdx.x;
    const int lane = tid & 31;
    const int wave = tid >> 5;
    const int hi = lane >> 4;
    const int ln = lane & 15;
    const long m0 = (long)blockIdx.x * 64 + wave * 16;
    const long n0 = (long)blockIdx.y * 64;

    v8f acc[4];
    #pragma unroll
    for (int t = 0; t < 4; ++t)
        #pragma unroll
        for (int i = 0; i < 8; ++i) acc[t][i] = 0.0f;

    const u16t* Arow = A + (m0 + ln) * lda;

    // stage a 64(N) x 32(K) tile of Bt into sB[buf]: 256 16-byte chunks,
    // 2 chunks per thread, via async global->LDS b128
    auto stageB = [&](int k, int buf) {
        #pragma unroll
        for (int c = tid; c < 256; c += 128) {
            int row = c >> 2, q = c & 3;
            async_copy16(Bt + (n0 + row) * (long)K + k + q * 8,
                         &sB[buf][row * BPITCH + q * 8]);
        }
    };

    stageB(0, 0);
    int cur = 0;
    for (int k = 0; k < K; k += 32) {
        const bool more = (k + 32) < K;
        if (more) {
            stageB(k + 32, cur ^ 1);
            async_wait<2>();       // current tile landed; next tile in flight
        } else {
            async_wait<0>();
        }
        __syncthreads();

        v16bf a = load_frag(Arow + k + hi * 8, Arow + k + 16 + hi * 8);
        #pragma unroll
        for (int t = 0; t < 4; ++t) {
            const u16t* bp = &sB[cur][(t * 16 + ln) * BPITCH + hi * 8];
            v16bf b = load_frag(bp, bp + 16);
            acc[t] = wmma_bf16(a, b, acc[t]);
        }
        __syncthreads();           // everyone done reading sB[cur] before overwrite
        cur ^= 1;
    }

    #pragma unroll
    for (int t = 0; t < 4; ++t) {
        long col = n0 + t * 16 + ln;
        float bv = bias ? bias[col] : 0.0f;
        #pragma unroll
        for (int r = 0; r < 8; ++r) {
            long row = m0 + r + 8 * hi;
            float v = acc[t][r] + bv;
            if (outF) {
                for (int rep = 0; rep < repCount; ++rep)
                    __builtin_nontemporal_store(v, &outF[row * ldcF + rep * repStride + col]);
            }
            if (outB) outB[row * ldcB + col] = f2bf(v);
        }
    }
}

// ---------------------------------------------------------------------------
// Attention: one wave per (n, h, 16-row L block).
//   scores(16x512) in LDS -> softmax -> probs bf16 -> O = P @ V via WMMA
//   Q rows: QKV[(l*8+n)*3072 + h*128 + d]
//   K rows: QKV[(m*8+n)*3072 + 1024 + h*128 + d]
//   V^T   : Vt[nh*65536 + d*512 + m]
//   O out : Obf[(l*8+n)*1024 + h*128 + d]  (bf16, feeds out-proj GEMM)
// ---------------------------------------------------------------------------
__global__ __launch_bounds__(32) void attn_wmma_kernel(
    const u16t* __restrict__ QKV,
    const u16t* __restrict__ Vt,
    u16t* __restrict__ Obf)
{
    __shared__ __align__(16) float sS[16 * Lrows];   // 32 KB
    __shared__ __align__(16) u16t  sP[16 * Lrows];   // 16 KB

    const int lane = threadIdx.x;
    const int hi = lane >> 4;
    const int ln = lane & 15;
    const int nh   = blockIdx.x >> 5;    // 0..63
    const int lblk = blockIdx.x & 31;    // 0..31
    const int n = nh >> 3, h = nh & 7;
    const int l0 = lblk * 16;
    const float scale = 0.08838834764831845f;   // 1/sqrt(128)

    const u16t* Qrow = QKV + ((long)(l0 + ln) * FREQ + n) * DIN + h * HD;

    // ---- phase 1: S = scale * Q K^T, stored to LDS ----
    for (int m0 = 0; m0 < Lrows; m0 += 16) {
        const u16t* Krow = QKV + ((long)(m0 + ln) * FREQ + n) * DIN + Edim + h * HD;
        v8f s;
        #pragma unroll
        for (int i = 0; i < 8; ++i) s[i] = 0.0f;
        #pragma unroll
        for (int kk = 0; kk < HD; kk += 32) {
            v16bf a = load_frag(Qrow + kk + hi * 8, Qrow + kk + 16 + hi * 8);
            v16bf b = load_frag(Krow + kk + hi * 8, Krow + kk + 16 + hi * 8);
            s = wmma_bf16(a, b, s);
        }
        #pragma unroll
        for (int r = 0; r < 8; ++r)
            sS[(r + 8 * hi) * Lrows + m0 + ln] = s[r] * scale;
    }
    __syncthreads();

    // ---- phase 2: softmax per row (lanes 0..15 handle one row each) ----
    if (lane < 16) {
        const int base = lane * Lrows;
        float mx = -3.0e38f;
        for (int m = 0; m < Lrows; ++m) mx = fmaxf(mx, sS[base + m]);
        float sum = 0.0f;
        for (int m = 0; m < Lrows; ++m) {
            float e = __expf(sS[base + m] - mx);
            sS[base + m] = e;
            sum += e;
        }
        float inv = 1.0f / sum;
        for (int m = 0; m < Lrows; ++m) sP[base + m] = f2bf(sS[base + m] * inv);
    }
    __syncthreads();

    // ---- phase 3: O = P @ V  (K = 512, N = 128 -> 8 D tiles) ----
    v8f o[8];
    #pragma unroll
    for (int t = 0; t < 8; ++t)
        #pragma unroll
        for (int i = 0; i < 8; ++i) o[t][i] = 0.0f;

    const u16t* Vbase = Vt + (long)nh * HD * Lrows;
    for (int kb = 0; kb < Lrows; kb += 32) {
        v16bf a = load_frag(&sP[ln * Lrows + kb + hi * 8],
                            &sP[ln * Lrows + kb + 16 + hi * 8]);
        #pragma unroll
        for (int t = 0; t < 8; ++t) {
            const u16t* vp = Vbase + (long)(t * 16 + ln) * Lrows + kb + hi * 8;
            v16bf b = load_frag(vp, vp + 16);
            o[t] = wmma_bf16(a, b, o[t]);
        }
    }

    #pragma unroll
    for (int t = 0; t < 8; ++t)
        #pragma unroll
        for (int r = 0; r < 8; ++r)
            Obf[((long)(l0 + r + 8 * hi) * FREQ + n) * Edim + h * HD + t * 16 + ln] =
                f2bf(o[t][r]);
}

// ---------------------------------------------------------------------------
// Host-side launch
// ---------------------------------------------------------------------------
extern "C" void kernel_launch(void* const* d_in, const int* in_sizes, int n_in,
                              void* d_out, int out_size, void* d_ws, size_t ws_size,
                              hipStream_t stream) {
    (void)in_sizes; (void)n_in; (void)out_size;

    const float* x    = (const float*)d_in[0];
    const float* encW = (const float*)d_in[1];
    const float* encB = (const float*)d_in[2];
    const float* decW = (const float*)d_in[3];
    const float* decB = (const float*)d_in[4];
    const float* Wi   = (const float*)d_in[5];
    const float* bi   = (const float*)d_in[6];
    const float* Wo   = (const float*)d_in[7];
    const float* bo   = (const float*)d_in[8];
    float* out = (float*)d_out;

    // workspace layout (bf16 = u16)
    char* ws = (char*)d_ws;
    size_t off = 0;
    auto alloc = [&](size_t elems) { u16t* p = (u16t*)(ws + off); off += elems * 2; return p; };
    u16t* Xbf    = alloc((size_t)MROWS * DIN);      // 4096x3072
    u16t* encWt  = alloc((size_t)Edim * DIN);       // 1024x3072 (W^T)
    u16t* Wibf   = alloc((size_t)3 * Edim * Edim);  // 3072x1024 (already N x K)
    u16t* Wobf   = alloc((size_t)Edim * Edim);      // 1024x1024
    u16t* decWt  = alloc((size_t)DIN * Edim);       // 3072x1024 (W^T)
    u16t* Zbf    = alloc((size_t)MROWS * Edim);     // 4096x1024
    u16t* QKVbf  = alloc((size_t)MROWS * DIN);      // 4096x3072
    u16t* Vt     = alloc((size_t)64 * HD * Lrows);  // 64x128x512
    u16t* Obf    = alloc((size_t)MROWS * Edim);     // 4096x1024
    u16t* Zrecbf = alloc((size_t)MROWS * Edim);     // 4096x1024
    if (off > ws_size) return;   // workspace too small; nothing safe to do

    const int CT = 256;
    auto cdiv = [](long a, long b) { return (int)((a + b - 1) / b); };

    // --- weight / activation converts ---
    convert_bf16_kernel<<<cdiv((long)MROWS * DIN, CT), CT, 0, stream>>>(x, Xbf, (long)MROWS * DIN);
    transpose_bf16_kernel<<<cdiv((long)DIN * Edim, CT), CT, 0, stream>>>(encW, encWt, DIN, Edim);
    convert_bf16_kernel<<<cdiv((long)3 * Edim * Edim, CT), CT, 0, stream>>>(Wi, Wibf, (long)3 * Edim * Edim);
    convert_bf16_kernel<<<cdiv((long)Edim * Edim, CT), CT, 0, stream>>>(Wo, Wobf, (long)Edim * Edim);
    transpose_bf16_kernel<<<cdiv((long)Edim * DIN, CT), CT, 0, stream>>>(decW, decWt, Edim, DIN);

    // --- encoder GEMM: Z = X * encW + b  -> Zbf ---
    gemm_bf16_wmma<<<dim3(MROWS / 64, Edim / 64), 128, 0, stream>>>(
        Xbf, (long)DIN, encWt, encB,
        nullptr, 0, 0, 0, Zbf, (long)Edim, DIN);

    // --- qkv GEMM: QKV = Z * Wi^T + bi -> QKVbf ---
    gemm_bf16_wmma<<<dim3(MROWS / 64, DIN / 64), 128, 0, stream>>>(
        Zbf, (long)Edim, Wibf, bi,
        nullptr, 0, 0, 0, QKVbf, (long)DIN, Edim);

    // --- V^T for attention ---
    build_vt_kernel<<<cdiv((long)64 * HD * Lrows, CT), CT, 0, stream>>>(QKVbf, Vt);

    // --- attention: 64 (n,h) pairs x 32 row-blocks, one wave each ---
    attn_wmma_kernel<<<64 * 32, 32, 0, stream>>>(QKVbf, Vt, Obf);

    // --- out projection: Zrec = O * Wo^T + bo -> Zrecbf ---
    gemm_bf16_wmma<<<dim3(MROWS / 64, Edim / 64), 128, 0, stream>>>(
        Obf, (long)Edim, Wobf, bo,
        nullptr, 0, 0, 0, Zrecbf, (long)Edim, Edim);

    // --- decoder GEMM: x_rec = Zrec * decW + b -> d_out[0 : 4096*3072] ---
    gemm_bf16_wmma<<<dim3(MROWS / 64, DIN / 64), 128, 0, stream>>>(
        Zrecbf, (long)Edim, decWt, decB,
        out, (long)DIN, 1, 0, nullptr, 0, Edim);

    // --- prediction: rows with f==7, decoded, replicated STEP times ---
    //     x_pred[b, s, :] = (Zrec[b*8+7] * decW + b) for s in 0..3
    float* outPred = out + (long)MROWS * DIN;   // after x_rec (512*384*64)
    gemm_bf16_wmma<<<dim3(Bz / 64, DIN / 64), 128, 0, stream>>>(
        Zrecbf + 7 * Edim, (long)(FREQ * Edim), decWt, decB,
        outPred, (long)(STEP * DIN), STEP, (long)DIN, nullptr, 0, Edim);
}
